// Frame_EGNN_68427418960434
// MI455X (gfx1250) — matrix operand order repaired and checked by reading, not compiled, not run on hardware
//
#include <hip/hip_runtime.h>

#define BMOL 1024
#define NNM 24
#define NNODE (BMOL * NNM)   // 24576
#define DEG (NNM - 1)        // 23 edges per node (row-sorted, contiguous)
#define EDGES (NNODE * DEG)  // 565248

typedef __attribute__((ext_vector_type(16))) _Float16 v16h;
typedef __attribute__((ext_vector_type(8)))  float    v8f;
typedef __attribute__((ext_vector_type(4)))  int      v4i;
typedef unsigned int u32;

// ---- CDNA5 feature detection (compile-safe fallbacks) ----------------------
#if defined(__has_builtin)
#if __has_builtin(__builtin_amdgcn_global_load_async_to_lds_b128)
#define HAVE_ASYNC_LDS 1
#endif
#if __has_builtin(__builtin_amdgcn_s_wait_asynccnt)
#define HAVE_WAIT_ASYNC 1
#endif
#if __has_builtin(__builtin_amdgcn_rcpf)
#define FRCP(x) __builtin_amdgcn_rcpf(x)
#endif
#endif
#ifndef FRCP
#define FRCP(x) (1.f / (x))
#endif

__device__ __forceinline__ float siluf(float x) { return x * FRCP(1.f + __expf(-x)); }
__device__ __forceinline__ float sigf(float x)  { return FRCP(1.f + __expf(-x)); }

__device__ __forceinline__ float wave_sum(float v) {
#pragma unroll
  for (int m = 16; m >= 1; m >>= 1) v += __shfl_xor(v, m, 32);
  return v;
}

__device__ __forceinline__ void unpack2(u32 w, float& a, float& b) {
  union { u32 u; _Float16 h[2]; } t; t.u = w; a = (float)t.h[0]; b = (float)t.h[1];
}
__device__ __forceinline__ u32 pack2(float a, float b) {
  union { u32 u; _Float16 h[2]; } t; t.h[0] = (_Float16)a; t.h[1] = (_Float16)b; return t.u;
}

// 16-byte global->LDS copy: async (ASYNCcnt-tracked) on CDNA5, VGPR bounce otherwise
#ifdef HAVE_ASYNC_LDS
typedef __attribute__((address_space(1))) v4i* gv4i_p;
typedef __attribute__((address_space(3))) v4i* lv4i_p;
#endif
__device__ __forceinline__ void async_cp16(const _Float16* gsrc, _Float16* ldst) {
#ifdef HAVE_ASYNC_LDS
  __builtin_amdgcn_global_load_async_to_lds_b128((gv4i_p)gsrc, (lv4i_p)ldst, 0, 0);
#else
  const u32* s4 = (const u32*)gsrc;
  u32* d4 = (u32*)ldst;
  d4[0] = s4[0]; d4[1] = s4[1]; d4[2] = s4[2]; d4[3] = s4[3];
#endif
}
__device__ __forceinline__ void wait_async0() {
#ifdef HAVE_ASYNC_LDS
#ifdef HAVE_WAIT_ASYNC
  __builtin_amdgcn_s_wait_asynccnt(0);
#else
  asm volatile("s_wait_asynccnt 0" ::: "memory");
#endif
#endif
}

// ---------------- WMMA fragment loaders (per ISA 7.12.2 layouts) -----------
__device__ __forceinline__ v16h load_afrag(const _Float16* A, int mt, int c, int kpad) {
  const int lane = threadIdx.x & 31;
  const int m  = mt * 16 + (lane & 15);
  const int kb = c * 32 + ((lane >> 4) << 3);
  const u32* p0 = (const u32*)(A + m * kpad + kb);
  const u32* p1 = (const u32*)(A + m * kpad + kb + 16);
  union { v16h h; u32 u[8]; } t;
#pragma unroll
  for (int i = 0; i < 4; ++i) { t.u[i] = p0[i]; t.u[4 + i] = p1[i]; }
  return t.h;
}

__device__ __forceinline__ v16h load_bfrag(const _Float16* Wn0, int c, int kpad) {
  const int lane = threadIdx.x & 31;
  const int n  = lane & 15;
  const int kb = c * 32 + ((lane >> 4) << 3);
  const u32* p0 = (const u32*)(Wn0 + n * kpad + kb);
  const u32* p1 = (const u32*)(Wn0 + n * kpad + kb + 16);
  union { v16h h; u32 u[8]; } t;
#pragma unroll
  for (int i = 0; i < 4; ++i) { t.u[i] = p0[i]; t.u[4 + i] = p1[i]; }
  return t.h;
}

template <int STRIDE, int ACT>
__device__ __forceinline__ void store_tile(_Float16* out, int n0, v8f acc, int mt) {
  const int lane = threadIdx.x & 31;
  const int n  = n0 + (lane & 15);
  const int mb = mt * 16 + ((lane >> 4) << 3);
#pragma unroll
  for (int r = 0; r < 8; ++r) {
    float v = acc[r];
    if (ACT == 1) v = siluf(v);
    out[(mb + r) * STRIDE + n] = (_Float16)v;
  }
}

constexpr int panel_n(int kpad, int nout) {
  int np = (8192 / kpad) / 16 * 16;
  if (np > nout) np = nout;
  while (nout % np) np -= 16;
  return np;
}

// Per-wave GEMM on pre-padded f16 weights Wp[NOUT][KPAD] (global).
template <int KPAD, int NOUT, int NVALID, int OSTRIDE, int ACT>
__device__ void gemm_stage(const _Float16* inb, _Float16* outb, _Float16* wt,
                           const _Float16* __restrict__ Wp, const float* __restrict__ bias) {
  constexpr int NP = panel_n(KPAD, NOUT);
  constexpr int CH = KPAD / 32;
  const int lane = threadIdx.x;
  for (int n0 = 0; n0 < NOUT; n0 += NP) {
    const _Float16* src = Wp + n0 * KPAD;
    for (int i = lane * 8; i < NP * KPAD; i += 256) async_cp16(src + i, wt + i);
    wait_async0();
    __syncthreads();
#pragma unroll 1
    for (int nt = 0; nt < NP / 16; ++nt) {
      const int ncol = n0 + nt * 16 + (lane & 15);
      const float bv = (bias != nullptr && ncol < NVALID) ? bias[ncol] : 0.f;
      v8f a0c = {bv, bv, bv, bv, bv, bv, bv, bv};
      v8f a1c = a0c;
#pragma unroll 1
      for (int c = 0; c < CH; ++c) {
        v16h bf  = load_bfrag(wt + nt * 16 * KPAD, c, KPAD);
        v16h fa0 = load_afrag(inb, 0, c, KPAD);
        v16h fa1 = load_afrag(inb, 1, c, KPAD);
        a0c = __builtin_amdgcn_wmma_f32_16x16x32_f16(false, fa0, false, bf, (short)0, a0c, false, false);
        a1c = __builtin_amdgcn_wmma_f32_16x16x32_f16(false, fa1, false, bf, (short)0, a1c, false, false);
      }
      store_tile<OSTRIDE, ACT>(outb, n0 + nt * 16, a0c, 0);
      store_tile<OSTRIDE, ACT>(outb, n0 + nt * 16, a1c, 1);
    }
    __syncthreads();
  }
}

// Row-wise LayerNorm (packed f16-pair LDS traffic); dim and stride are even.
__device__ void ln_rows(_Float16* buf, int stride, int dim,
                        const float* __restrict__ g, const float* __restrict__ b) {
  const int lane = threadIdx.x;
  __syncthreads();
  u32* row = (u32*)(buf + lane * stride);
  const int d2 = dim >> 1;
  float s = 0.f, s2 = 0.f;
  for (int k = 0; k < d2; ++k) {
    float v0, v1; unpack2(row[k], v0, v1);
    s += v0 + v1; s2 += v0 * v0 + v1 * v1;
  }
  float mu = s / (float)dim;
  float rs = rsqrtf(s2 / (float)dim - mu * mu + 1e-5f);
  for (int k = 0; k < d2; ++k) {
    float v0, v1; unpack2(row[k], v0, v1);
    row[k] = pack2((v0 - mu) * rs * g[2 * k] + b[2 * k],
                   (v1 - mu) * rs * g[2 * k + 1] + b[2 * k + 1]);
  }
  __syncthreads();
}

// ---------------- equivariant local frame ----------------------------------
struct Frame { float xd[3], xc[3], xv[3], fi[3], fj[3], fr, cs, sn, radial; };

__device__ void build_frame(const float* __restrict__ x, int row, int col, Frame& F) {
  float xr[3], xcl[3];
#pragma unroll
  for (int d = 0; d < 3; ++d) { xr[d] = x[row * 3 + d]; xcl[d] = x[col * 3 + d]; }
  float dx[3] = {xr[0] - xcl[0], xr[1] - xcl[1], xr[2] - xcl[2]};
  float ss = dx[0] * dx[0] + dx[1] * dx[1] + dx[2] * dx[2];
  F.radial = sqrtf(ss + 1e-8f);
  float cr[3] = {xr[1] * xcl[2] - xr[2] * xcl[1],
                 xr[2] * xcl[0] - xr[0] * xcl[2],
                 xr[0] * xcl[1] - xr[1] * xcl[0]};
  float nd = sqrtf(ss + 3e-30f);
  float nc = sqrtf(cr[0] * cr[0] + cr[1] * cr[1] + cr[2] * cr[2] + 3e-30f);
#pragma unroll
  for (int d = 0; d < 3; ++d) { F.xd[d] = dx[d] / nd; F.xc[d] = cr[d] / nc; }
  F.xv[0] = F.xd[1] * F.xc[2] - F.xd[2] * F.xc[1];
  F.xv[1] = F.xd[2] * F.xc[0] - F.xd[0] * F.xc[2];
  F.xv[2] = F.xd[0] * F.xc[1] - F.xd[1] * F.xc[0];
  F.fi[0] = xr[0] * F.xd[0] + xr[1] * F.xd[1] + xr[2] * F.xd[2];
  F.fi[1] = xr[0] * F.xc[0] + xr[1] * F.xc[1] + xr[2] * F.xc[2];
  F.fi[2] = xr[0] * F.xv[0] + xr[1] * F.xv[1] + xr[2] * F.xv[2];
  F.fj[0] = xcl[0] * F.xd[0] + xcl[1] * F.xd[1] + xcl[2] * F.xd[2];
  F.fj[1] = xcl[0] * F.xc[0] + xcl[1] * F.xc[1] + xcl[2] * F.xc[2];
  F.fj[2] = xcl[0] * F.xv[0] + xcl[1] * F.xv[1] + xcl[2] * F.xv[2];
  float d0 = F.fi[0] - F.fj[0], d1 = F.fi[1] - F.fj[1], d2 = F.fi[2] - F.fj[2];
  F.fr = d0 * d0 + d1 * d1 + d2 * d2;
  float ni = fmaxf(sqrtf(F.fi[0] * F.fi[0] + F.fi[1] * F.fi[1] + F.fi[2] * F.fi[2]), 1e-8f);
  float nj = fmaxf(sqrtf(F.fj[0] * F.fj[0] + F.fj[1] * F.fj[1] + F.fj[2] * F.fj[2]), 1e-8f);
  float c = (F.fi[0] * F.fj[0] + F.fi[1] * F.fj[1] + F.fi[2] * F.fj[2]) / (ni * nj);
  c = fminf(fmaxf(c, -1.f), 1.f);
  F.cs = c; F.sn = sqrtf(1.f - c * c + 1e-30f);
}

// ================= parameter bundles ========================================
struct GclP {
  const _Float16 *el1p, *el2p, *bl1p, *bl2p;                 // pre-padded f16 GEMM weights
  const float *attW, *attB, *el1B, *el2B, *bl1B, *bl2B,
              *bln1b, *bln1g, *bln2b, *bln2g, *eln1b, *eln1g, *eln2b, *eln2g,
              *nl1W, *nl1B, *nl2W, *nl2B, *nlnb, *nlng;
};
struct EnP {
  const _Float16 *m1p, *m2p;
  const float *m1B, *m2B, *ln1b, *ln1g, *ln2b, *ln2g,
              *nehW, *nehB, *attW, *attB, *p1W, *p1B, *p2W, *plnb, *plng;
};
struct FoP {
  const _Float16 *f1p, *f2p;
  const float *f1B, *flnb, *flng;
};

// ================= prep / small kernels =====================================
__global__ void k_pad_w(const float* __restrict__ W, _Float16* __restrict__ out,
                        int NOUT, int KPAD, int NVALID, int KIN) {
  int i = blockIdx.x * blockDim.x + threadIdx.x;
  if (i >= NOUT * KPAD) return;
  int n = i / KPAD, k = i - n * KPAD;
  float v = (n < NVALID && k < KIN) ? W[n * KIN + k] : 0.f;
  out[i] = (_Float16)v;
}

__global__ void k_node_emb(const float* __restrict__ h, const float* __restrict__ W,
                           const float* __restrict__ b, float* __restrict__ out) {
  int n = blockIdx.x * blockDim.x + threadIdx.x;
  if (n >= NNODE) return;
#pragma unroll
  for (int o = 0; o < 8; ++o) {
    float s = b[o];
#pragma unroll
    for (int k = 0; k < 16; ++k) s += h[n * 16 + k] * W[o * 16 + k];
    out[n * 8 + o] = s;
  }
}

__global__ void k_bond_emb(const float* __restrict__ bond, const float* __restrict__ W,
                           const float* __restrict__ b, float* __restrict__ out) {
  int e = blockIdx.x * blockDim.x + threadIdx.x;
  if (e >= EDGES) return;
#pragma unroll
  for (int o = 0; o < 4; ++o) {
    float s = b[o];
#pragma unroll
    for (int k = 0; k < 8; ++k) s += bond[e * 8 + k] * W[o * 8 + k];
    out[e * 4 + o] = s;
  }
}

__global__ void k_pool(const float* __restrict__ e_node, const float* __restrict__ W,
                       float* __restrict__ out) {
  int b = blockIdx.x * blockDim.x + threadIdx.x;
  if (b >= BMOL) return;
  float s = 0.f;
#pragma unroll
  for (int n = 0; n < NNM; ++n) s += W[n] * e_node[b * NNM + n];
  out[b] = siluf(s);
}

// ================= main fused kernels =======================================
__global__ __launch_bounds__(32) void k_gcl(const float* __restrict__ x,
    const float* __restrict__ h_in, const float* __restrict__ bond_in,
    const int* __restrict__ colIdx, float* __restrict__ h_out,
    float* __restrict__ bond_out, GclP P) {
  __shared__ _Float16 wt[8192];
  __shared__ _Float16 actA[32 * 256];
  __shared__ _Float16 actB[32 * 256];
  __shared__ float sbuf[256];
  const int node = blockIdx.x;
  const int lane = threadIdx.x;
  const int eb = node * DEG;

  { // X0 = [h_i(8), h_col(8), bond(4), rbf(16)] -> pad 64
    _Float16* rowp = actA + lane * 64;
    if (lane < DEG) {
      const int e = eb + lane;
      const int col = colIdx[e];
#pragma unroll
      for (int k = 0; k < 8; ++k) { rowp[k] = (_Float16)h_in[node * 8 + k]; rowp[8 + k] = (_Float16)h_in[col * 8 + k]; }
#pragma unroll
      for (int k = 0; k < 4; ++k) rowp[16 + k] = (_Float16)bond_in[e * 4 + k];
      float dx = x[node * 3 + 0] - x[col * 3 + 0];
      float dy = x[node * 3 + 1] - x[col * 3 + 1];
      float dz = x[node * 3 + 2] - x[col * 3 + 2];
      float r = sqrtf(dx * dx + dy * dy + dz * dz + 1e-8f);
#pragma unroll
      for (int k = 0; k < 16; ++k) { float d = r - (float)k * (1.f / 3.f); rowp[20 + k] = (_Float16)__expf(-4.5f * d * d); }
      for (int k = 18; k < 32; ++k) ((u32*)rowp)[k] = 0u;
    } else {
      for (int k = 0; k < 32; ++k) ((u32*)rowp)[k] = 0u;
    }
  }
  ln_rows(actA, 64, 36, P.eln1g, P.eln1b);
  gemm_stage<64, 256, 256, 256, 1>(actA, actB, wt, P.el1p, P.el1B);
  ln_rows(actB, 256, 256, P.eln2g, P.eln2b);
  gemm_stage<256, 128, 128, 128, 1>(actB, actA, wt, P.el2p, P.el2B);

  __syncthreads();
  { // attention gate per edge row (packed-pair LDS access)
    u32* row = (u32*)(actA + lane * 128);
    float a = P.attB[0];
    for (int k = 0; k < 64; ++k) {
      float v0, v1; unpack2(row[k], v0, v1);
      a += v0 * P.attW[2 * k] + v1 * P.attW[2 * k + 1];
    }
    float gv = sigf(a);
    for (int k = 0; k < 64; ++k) {
      float v0, v1; unpack2(row[k], v0, v1);
      row[k] = pack2(v0 * gv, v1 * gv);
    }
  }
  __syncthreads();
  { // local segment-sum over this node's 23 edges (4 cols per lane)
    float s0 = 0.f, s1 = 0.f, s2 = 0.f, s3 = 0.f;
    for (int e = 0; e < DEG; ++e) {
      const u32* rp = (const u32*)(actA + e * 128 + lane * 4);
      float a0, a1, a2, a3;
      unpack2(rp[0], a0, a1); unpack2(rp[1], a2, a3);
      s0 += a0; s1 += a1; s2 += a2; s3 += a3;
    }
    sbuf[lane * 4 + 0] = s0; sbuf[lane * 4 + 1] = s1;
    sbuf[lane * 4 + 2] = s2; sbuf[lane * 4 + 3] = s3;
  }
  __syncthreads();
  { // bond branch X = [bond(4), m(128)] -> pad 160
    _Float16* rowp = actB + lane * 160;
    if (lane < DEG) {
      const int e = eb + lane;
#pragma unroll
      for (int k = 0; k < 4; ++k) rowp[k] = (_Float16)bond_in[e * 4 + k];
      const u32* src = (const u32*)(actA + lane * 128);
      u32* dst = (u32*)(rowp + 4);
      for (int k = 0; k < 64; ++k) dst[k] = src[k];
      for (int k = 66; k < 80; ++k) ((u32*)rowp)[k] = 0u;
    } else {
      for (int k = 0; k < 80; ++k) ((u32*)rowp)[k] = 0u;
    }
  }
  ln_rows(actB, 160, 132, P.bln1g, P.bln1b);
  gemm_stage<160, 128, 128, 128, 1>(actB, actA, wt, P.bl1p, P.bl1B);
  ln_rows(actA, 128, 128, P.bln2b ? P.bln2g : P.bln2g, P.bln2b);
  gemm_stage<128, 16, 4, 16, 1>(actA, actB, wt, P.bl2p, P.bl2B);
  __syncthreads();
  if (lane < DEG) {
#pragma unroll
    for (int q = 0; q < 4; ++q) bond_out[(eb + lane) * 4 + q] = (float)actB[lane * 16 + q];
  }
  { // node update: hn = n_l2(ln(silu(n_l1([h, agg]))))
    float hv[8];
#pragma unroll
    for (int k = 0; k < 8; ++k) hv[k] = h_in[node * 8 + k];
    float t[4];
#pragma unroll
    for (int j = 0; j < 4; ++j) {
      int o = lane + 32 * j;
      float s = P.nl1B[o];
      const float* wr = P.nl1W + o * 136;
#pragma unroll
      for (int k = 0; k < 8; ++k) s += hv[k] * wr[k];
      for (int k = 0; k < 128; ++k) s += sbuf[k] * wr[8 + k];
      t[j] = siluf(s);
    }
    float ls = t[0] + t[1] + t[2] + t[3];
    float ls2 = t[0] * t[0] + t[1] * t[1] + t[2] * t[2] + t[3] * t[3];
    float S = wave_sum(ls), S2 = wave_sum(ls2);
    float mu = S / 128.f, rs = rsqrtf(S2 / 128.f - mu * mu + 1e-5f);
    __syncthreads();
#pragma unroll
    for (int j = 0; j < 4; ++j) { int o = lane + 32 * j; sbuf[128 + o] = (t[j] - mu) * rs * P.nlng[o] + P.nlnb[o]; }
    __syncthreads();
    if (lane < 8) {
      float s = P.nl2B[lane];
      const float* wr = P.nl2W + lane * 128;
      for (int k = 0; k < 128; ++k) s += sbuf[128 + k] * wr[k];
      h_out[node * 8 + lane] = s;
    }
  }
}

__global__ __launch_bounds__(32) void k_energy(const float* __restrict__ x,
    const float* __restrict__ h_in, const float* __restrict__ bond_in,
    const int* __restrict__ colIdx, EnP P, float* __restrict__ neh_out,
    float* __restrict__ e_node) {
  __shared__ _Float16 wt[8192];
  __shared__ _Float16 actA[32 * 256];
  __shared__ _Float16 actB[32 * 256];
  __shared__ float sbuf[256];
  const int node = blockIdx.x;
  const int lane = threadIdx.x;
  const int eb = node * DEG;

  { // X = [h_i(8), h_col(8), ff(9), bond(4), rbf(16)] = 45 -> pad 64 (raw)
    _Float16* rowp = actA + lane * 64;
    if (lane < DEG) {
      const int e = eb + lane;
      const int col = colIdx[e];
      Frame F; build_frame(x, node, col, F);
#pragma unroll
      for (int k = 0; k < 8; ++k) { rowp[k] = (_Float16)h_in[node * 8 + k]; rowp[8 + k] = (_Float16)h_in[col * 8 + k]; }
#pragma unroll
      for (int d = 0; d < 3; ++d) { rowp[16 + d] = (_Float16)F.fi[d]; rowp[19 + d] = (_Float16)F.fj[d]; }
      rowp[22] = (_Float16)F.fr; rowp[23] = (_Float16)F.cs; rowp[24] = (_Float16)F.sn;
#pragma unroll
      for (int k = 0; k < 4; ++k) rowp[25 + k] = (_Float16)bond_in[e * 4 + k];
#pragma unroll
      for (int k = 0; k < 16; ++k) { float d = F.radial - (float)k * (1.f / 3.f); rowp[29 + k] = (_Float16)__expf(-4.5f * d * d); }
      for (int k = 45; k < 64; ++k) rowp[k] = (_Float16)0.f;
    } else {
      for (int k = 0; k < 32; ++k) ((u32*)rowp)[k] = 0u;
    }
  }
  __syncthreads();
  gemm_stage<64, 256, 256, 256, 1>(actA, actB, wt, P.m1p, P.m1B);
  ln_rows(actB, 256, 256, P.ln1g, P.ln1b);
  gemm_stage<256, 128, 128, 128, 1>(actB, actA, wt, P.m2p, P.m2B);
  ln_rows(actA, 128, 128, P.ln2g, P.ln2b);
  { // segment sum
    float s0 = 0.f, s1 = 0.f, s2 = 0.f, s3 = 0.f;
    for (int e = 0; e < DEG; ++e) {
      const u32* rp = (const u32*)(actA + e * 128 + lane * 4);
      float a0, a1, a2, a3;
      unpack2(rp[0], a0, a1); unpack2(rp[1], a2, a3);
      s0 += a0; s1 += a1; s2 += a2; s3 += a3;
    }
    sbuf[lane * 4 + 0] = s0; sbuf[lane * 4 + 1] = s1;
    sbuf[lane * 4 + 2] = s2; sbuf[lane * 4 + 3] = s3;
  }
  __syncthreads();
  { // neh + attention + per-node energy head
    float hv[8];
#pragma unroll
    for (int k = 0; k < 8; ++k) hv[k] = h_in[node * 8 + k];
    float t[4];
#pragma unroll
    for (int j = 0; j < 4; ++j) {
      int o = lane + 32 * j;
      float s = P.nehB[o];
      const float* wr = P.nehW + o * 136;
#pragma unroll
      for (int k = 0; k < 8; ++k) s += hv[k] * wr[k];
      for (int k = 0; k < 128; ++k) s += sbuf[k] * wr[8 + k];
      t[j] = siluf(s);
      neh_out[node * 128 + o] = t[j];
    }
    __syncthreads();
#pragma unroll
    for (int j = 0; j < 4; ++j) sbuf[128 + lane + 32 * j] = t[j];
    __syncthreads();
    float pa = t[0] * P.attW[lane] + t[1] * P.attW[lane + 32] +
               t[2] * P.attW[lane + 64] + t[3] * P.attW[lane + 96];
    float gv = sigf(wave_sum(pa) + P.attB[0]);
    float u[4];
#pragma unroll
    for (int j = 0; j < 4; ++j) {
      int o = lane + 32 * j;
      float s = P.p1B[o];
      const float* wr = P.p1W + o * 128;
      for (int k = 0; k < 128; ++k) s += gv * sbuf[128 + k] * wr[k];
      u[j] = siluf(s);
    }
    float ls = u[0] + u[1] + u[2] + u[3];
    float ls2 = u[0] * u[0] + u[1] * u[1] + u[2] * u[2] + u[3] * u[3];
    float S = wave_sum(ls), S2 = wave_sum(ls2);
    float mu = S / 128.f, rs = rsqrtf(S2 / 128.f - mu * mu + 1e-5f);
    float pe = 0.f;
#pragma unroll
    for (int j = 0; j < 4; ++j) {
      int o = lane + 32 * j;
      float tn = (u[j] - mu) * rs * P.plng[o] + P.plnb[o];
      pe += tn * P.p2W[o];
    }
    float Ev = wave_sum(pe);
    if (lane == 0) e_node[node] = Ev;
  }
}

__global__ __launch_bounds__(32) void k_force(const float* __restrict__ x,
    const float* __restrict__ neh, const int* __restrict__ colIdx, FoP P,
    float* __restrict__ out_force) {
  __shared__ _Float16 wt[8192];
  __shared__ _Float16 actA[32 * 288];
  __shared__ _Float16 actB[32 * 128];
  __shared__ float sbuf[128];
  const int node = blockIdx.x;
  const int lane = threadIdx.x;
  const int eb = node * DEG;
  const bool valid = lane < DEG;
  Frame F = {};
  { // X = [neh_row(128), neh_col(128), fi(3), fj(3)] = 262 -> pad 288
    _Float16* rowp = actA + lane * 288;
    if (valid) {
      const int e = eb + lane;
      const int col = colIdx[e];
      build_frame(x, node, col, F);
      u32* dst = (u32*)rowp;
      const float* nr = neh + node * 128;
      const float* ncv = neh + col * 128;
      for (int k = 0; k < 64; ++k) dst[k] = pack2(nr[2 * k], nr[2 * k + 1]);
      for (int k = 0; k < 64; ++k) dst[64 + k] = pack2(ncv[2 * k], ncv[2 * k + 1]);
#pragma unroll
      for (int d = 0; d < 3; ++d) { rowp[256 + d] = (_Float16)F.fi[d]; rowp[259 + d] = (_Float16)F.fj[d]; }
      for (int k = 131; k < 144; ++k) dst[k] = 0u;
    } else {
      u32* dst = (u32*)rowp;
      for (int k = 0; k < 144; ++k) dst[k] = 0u;
    }
  }
  __syncthreads();
  gemm_stage<288, 128, 128, 128, 1>(actA, actB, wt, P.f1p, P.f1B);
  ln_rows(actB, 128, 128, P.flng, P.flnb);
  gemm_stage<128, 16, 3, 16, 0>(actB, actA, wt, P.f2p, nullptr);
  __syncthreads();
  if (valid) {
    float f0 = (float)actA[lane * 16 + 0];
    float f1 = (float)actA[lane * 16 + 1];
    float f2 = (float)actA[lane * 16 + 2];
#pragma unroll
    for (int d = 0; d < 3; ++d) {
      float ef = f0 * F.xd[d] + f1 * F.xc[d] + f2 * F.xv[d];
      ef = fminf(fmaxf(ef, -1000.f), 1000.f);
      sbuf[lane * 3 + d] = ef;
    }
  }
  __syncthreads();
  if (lane < 3) {
    float s = 0.f;
    for (int e = 0; e < DEG; ++e) s += sbuf[e * 3 + lane];
    out_force[node * 3 + lane] = s / (float)DEG;
  }
}

// ================= host launcher ===========================================
extern "C" void kernel_launch(void* const* d_in, const int* in_sizes, int n_in,
                              void* d_out, int out_size, void* d_ws, size_t ws_size,
                              hipStream_t stream) {
  (void)in_sizes; (void)n_in; (void)out_size; (void)ws_size;
  auto F = [&](int i) { return (const float*)d_in[i]; };
  const float* x     = F(0);
  const float* h0    = F(1);
  const float* bond0 = F(2);
  const int* ei  = (const int*)d_in[3];
  const int* col = ei + EDGES;

  float* ws = (float*)d_ws;
  float* bondA = ws; ws += (size_t)EDGES * 4;
  float* bondB = ws; ws += (size_t)EDGES * 4;
  float* hA    = ws; ws += (size_t)NNODE * 8;
  float* hB    = ws; ws += (size_t)NNODE * 8;
  float* neh   = ws; ws += (size_t)NNODE * 128;
  float* enode = ws; ws += (size_t)NNODE;

  _Float16* hp = (_Float16*)ws;
  auto carve = [&](size_t n) { _Float16* p = hp; hp += n; return p; };
  _Float16* el1p[3]; _Float16* el2p[3]; _Float16* bl1p[3]; _Float16* bl2p[3];
  for (int L = 0; L < 3; ++L) {
    el1p[L] = carve(256 * 64);
    el2p[L] = carve(128 * 256);
    bl1p[L] = carve(128 * 160);
    bl2p[L] = carve(16 * 128);
  }
  _Float16* m1p = carve(256 * 64);
  _Float16* m2p = carve(128 * 256);
  _Float16* f1p = carve(128 * 288);
  _Float16* f2p = carve(16 * 128);

  auto pad = [&](const float* W, _Float16* out, int NOUT, int KPAD, int NVALID, int KIN) {
    int sz = NOUT * KPAD;
    k_pad_w<<<(sz + 255) / 256, 256, 0, stream>>>(W, out, NOUT, KPAD, NVALID, KIN);
  };
  for (int L = 0; L < 3; ++L) {
    int g = 29 + L * 24;
    pad(F(g + 10), el1p[L], 256, 64, 256, 36);
    pad(F(g + 12), el2p[L], 128, 256, 128, 256);
    pad(F(g + 2),  bl1p[L], 128, 160, 128, 132);
    pad(F(g + 4),  bl2p[L], 16, 128, 4, 128);
  }
  pad(F(8),  m1p, 256, 64, 256, 45);
  pad(F(10), m2p, 128, 256, 128, 256);
  pad(F(24), f1p, 128, 288, 128, 262);
  pad(F(26), f2p, 16, 128, 3, 128);

  k_node_emb<<<(NNODE + 255) / 256, 256, 0, stream>>>(h0, F(101), F(102), hA);
  k_bond_emb<<<(EDGES + 255) / 256, 256, 0, stream>>>(bond0, F(4), F(5), bondA);

  const float* hin = hA;   float* hout = hB;
  const float* bin = bondA; float* bout = bondB;
  for (int L = 0; L < 3; ++L) {
    int g = 29 + L * 24;
    GclP P;
    P.el1p = el1p[L]; P.el2p = el2p[L]; P.bl1p = bl1p[L]; P.bl2p = bl2p[L];
    P.attW = F(g + 0);  P.attB = F(g + 1);
    P.bl1B = F(g + 3);  P.bl2B = F(g + 5);
    P.bln1b = F(g + 6); P.bln1g = F(g + 7); P.bln2b = F(g + 8); P.bln2g = F(g + 9);
    P.el1B = F(g + 11); P.el2B = F(g + 13);
    P.eln1b = F(g + 14); P.eln1g = F(g + 15); P.eln2b = F(g + 16); P.eln2g = F(g + 17);
    P.nl1W = F(g + 18); P.nl1B = F(g + 19); P.nl2W = F(g + 20); P.nl2B = F(g + 21);
    P.nlnb = F(g + 22); P.nlng = F(g + 23);
    k_gcl<<<NNODE, 32, 0, stream>>>(x, hin, bin, col, hout, bout, P);
    const float* th = hin; hin = hout; hout = (float*)th;
    const float* tb = bin; bin = bout; bout = (float*)tb;
  }

  EnP EP;
  EP.m1p = m1p; EP.m2p = m2p;
  EP.m1B = F(9); EP.m2B = F(11);
  EP.ln1b = F(12); EP.ln1g = F(13); EP.ln2b = F(14); EP.ln2g = F(15);
  EP.nehW = F(16); EP.nehB = F(17); EP.attW = F(6); EP.attB = F(7);
  EP.p1W = F(18); EP.p1B = F(19); EP.p2W = F(20); EP.plnb = F(21); EP.plng = F(22);
  k_energy<<<NNODE, 32, 0, stream>>>(x, hin, bin, col, EP, neh, enode);
  k_pool<<<(BMOL + 255) / 256, 256, 0, stream>>>(enode, F(23), (float*)d_out);

  FoP FP;
  FP.f1p = f1p; FP.f2p = f2p;
  FP.f1B = F(25); FP.flnb = F(27); FP.flng = F(28);
  k_force<<<NNODE, 32, 0, stream>>>(x, neh, col, FP, (float*)d_out + BMOL);
}
